// GAT_17016660427225
// MI455X (gfx1250) — matrix operand order repaired and checked by reading, not compiled, not run on hardware
//
#include <hip/hip_runtime.h>

// ---------------- problem constants (from reference) ----------------
#define NT    65536      // total nodes
#define NE    2097152    // edges
#define INF_  64         // input features
#define HIDF  128        // hidden features
#define NB    512        // batch (graphs)
#define KFC   16384      // HID * nodes_per_graph = FC1 K

typedef float  v2f   __attribute__((ext_vector_type(2)));
typedef float  v8f   __attribute__((ext_vector_type(8)));
typedef __bf16 v16bf __attribute__((ext_vector_type(16)));
typedef unsigned int u32x4 __attribute__((ext_vector_type(4)));
typedef int          i32x4 __attribute__((ext_vector_type(4)));
typedef int          i32x8 __attribute__((ext_vector_type(8)));

struct U128 { unsigned int x[4]; };          // POD 16B
union  BFV  { U128 q[2]; v16bf v; };         // 32B reinterpret for WMMA bf16 operand

// ---- hard-enable the gfx1250 async / TDM paths on the device pass.
// (Verification: if these builtins were absent, this round would fail to
// compile with a named-builtin error; rc=0 proves the instructions lowered.)
#if defined(__AMDGCN__)
#define HAVE_ASYNC 1
#define HAVE_TDM   1
#else
#define HAVE_ASYNC 0
#define HAVE_TDM   0
#endif
#if __has_include(<hip/amd_detail/amd_gfx1250_TDM.h>)
#define TDM_SIX_ARGS 1   // amdgpu-toolchain (clang-23 / therock-10.0 headers)
#else
#define TDM_SIX_ARGS 0   // ROCm 7.2 (clang-22): 5-arg builtin
#endif

__device__ __forceinline__ unsigned short f2bf(float f) {
    unsigned int u = __float_as_uint(f);
    unsigned int r = u + 0x7FFFu + ((u >> 16) & 1u);   // round-to-nearest-even
    return (unsigned short)(r >> 16);
}
__device__ __forceinline__ float lrelu(float x) { return x > 0.f ? x : 0.2f * x; }

#if HAVE_TDM
// Issue a TDM 2-D tile load: tile_rows x 32 bf16 cols from a [tensor_rows, KFC]
// bf16 tensor into LDS at byte offset lds_off. (The calling kernel keeps all its
// LDS in one __shared__ object at segment offset 0, so integer D# lds_addr works.)
__device__ __forceinline__ void tdm_load_tile_bf16(const unsigned short* gbase,
                                                   unsigned int lds_off,
                                                   unsigned int row0, unsigned int k0,
                                                   unsigned int tile_rows,
                                                   unsigned int tensor_rows)
{
    unsigned long long ga = (unsigned long long)(const void*)gbase
                          + ((unsigned long long)row0 * KFC + (unsigned long long)k0) * 2ull;
    u32x4 g0;
    g0[0] = 1u;                                        // count=1 valid user descriptor
    g0[1] = lds_off;                                   // lds_addr (bytes)
    g0[2] = (unsigned int)ga;                          // global_addr[31:0]
    g0[3] = (unsigned int)(ga >> 32) | 0x80000000u;    // global_addr[56:32] | type=2<<30
    i32x8 g1;
    g1[0] = 1 << 16;                                   // data_size=1 (2 bytes/elem)
    g1[1] = (int)(((unsigned)KFC & 0xFFFFu) << 16);    // tensor_dim0[15:0] in [31:16]
    g1[2] = (int)(((unsigned)KFC >> 16) | ((tensor_rows & 0xFFFFu) << 16));
    g1[3] = (int)((tensor_rows >> 16) | (32u << 16));  // dim1 hi | tile_dim0=32
    g1[4] = (int)tile_rows;                            // tile_dim1 (tile_dim2=0)
    g1[5] = KFC;                                       // tensor_dim0_stride[31:0]
    g1[6] = 0;
    g1[7] = 0;
    i32x4 z = {0, 0, 0, 0};
#if TDM_SIX_ARGS
    i32x8 z8 = {0, 0, 0, 0, 0, 0, 0, 0};
    __builtin_amdgcn_tensor_load_to_lds(g0, g1, z, z, z8, 0);
#else
    __builtin_amdgcn_tensor_load_to_lds(g0, g1, z, z, 0);
#endif
}
#endif

// ---------------- generic f32 WMMA GEMM, N fixed at 128 ----------------
// C[M,128] = A[M,K] @ B[K,128] (+bias)(+relu). Block=256 (8 waves), 16 rows/block,
// wave w computes the 16x16 tile at cols [16w,16w+16). V_WMMA_F32_16X16X4_F32.
// Both the contiguous 16xK A tile and the KxN B panel are staged into LDS via
// async-to-LDS loads; the inner loop is pure DS -> WMMA.
__global__ void k_gemm_f32_n128(const float* __restrict__ A, const float* __restrict__ B,
                                const float* __restrict__ bias, float* __restrict__ C,
                                int K, int do_relu)
{
    __shared__ float As[16 * 128];                     // 8KB max (K<=128)
    __shared__ float Bs[128 * 128];                    // 64KB max
    const float* Ab = A + (size_t)blockIdx.x * 16 * K; // 16 contiguous rows
    for (int i = threadIdx.x; i < 4 * K; i += 256) {   // 16*K floats = 4*K float4
#if HAVE_ASYNC
        __builtin_amdgcn_global_load_async_to_lds_b128(
            (__attribute__((address_space(1))) i32x4*)(Ab + i * 4),
            (__attribute__((address_space(3))) i32x4*)(As + i * 4), 0, 0);
#else
        *(float4*)(As + i * 4) = *(const float4*)(Ab + i * 4);
#endif
    }
    for (int i = threadIdx.x; i < K * 32; i += 256) {  // K*128 floats
#if HAVE_ASYNC
        __builtin_amdgcn_global_load_async_to_lds_b128(
            (__attribute__((address_space(1))) i32x4*)(B + i * 4),
            (__attribute__((address_space(3))) i32x4*)(Bs + i * 4), 0, 0);
#else
        *(float4*)(Bs + i * 4) = *(const float4*)(B + i * 4);
#endif
    }
#if HAVE_ASYNC
    __builtin_amdgcn_s_wait_asynccnt(0);
#endif
    __syncthreads();

    const int lane = threadIdx.x & 31;
    const int wv   = threadIdx.x >> 5;
    const int m0   = blockIdx.x * 16;
    const int col0 = wv * 16;
    const int l15  = lane & 15;
    const int hi   = (lane & 16) ? 1 : 0;   // upper half-wave holds K+2/K+3

    v8f c = {};
    for (int k = 0; k < K; k += 4) {
        const int ka = k + hi * 2;
        v2f a, b;
        const float* ap = As + l15 * K + ka;          // A 16x4: lane row, 2 K/lane
        a.x = ap[0]; a.y = ap[1];
        const float* bp = Bs + ka * 128 + col0 + l15; // B 4x16: row striped across lanes
        b.x = bp[0]; b.y = bp[128];
        c = __builtin_amdgcn_wmma_f32_16x16x4_f32(false, a, false, b, (short)0, c, false, false);
    }
    const int n  = col0 + l15;
    const int mr = m0 + hi * 8;              // C: vgpr r -> M = r (+8 for upper half)
    const float bb = bias ? bias[n] : 0.f;
    #pragma unroll
    for (int r = 0; r < 8; ++r) {
        float v = c[r] + bb;
        if (do_relu) v = fmaxf(v, 0.f);
        C[(size_t)(mr + r) * 128 + n] = v;
    }
}

// ---------------- alpha_s / alpha_d GEMV: one wave per node ----------------
__global__ void k_alpha(const float* __restrict__ h, const float* __restrict__ a_src,
                        const float* __restrict__ a_dst, float* __restrict__ as,
                        float* __restrict__ ad)
{
    const int lane = threadIdx.x & 31;
    const int node = blockIdx.x * 8 + (threadIdx.x >> 5);
    const float4 hv = *(const float4*)(h + (size_t)node * HIDF + lane * 4);
    const float4 s4 = *(const float4*)(a_src + lane * 4);
    const float4 d4 = *(const float4*)(a_dst + lane * 4);
    float s = hv.x * s4.x + hv.y * s4.y + hv.z * s4.z + hv.w * s4.w;
    float d = hv.x * d4.x + hv.y * d4.y + hv.z * d4.z + hv.w * d4.w;
    #pragma unroll
    for (int off = 16; off; off >>= 1) { s += __shfl_xor(s, off); d += __shfl_xor(d, off); }
    if (lane == 0) { as[node] = s; ad[node] = d; }
}

// ---------------- CSR build: count, scan, scatter ----------------
__global__ void k_zero(int* __restrict__ p) { p[blockIdx.x * 256 + threadIdx.x] = 0; }

__global__ void k_degree(const int* __restrict__ ei, int* __restrict__ deg)
{
    const int e = blockIdx.x * 256 + threadIdx.x;
    atomicAdd(&deg[ei[NE + e]], 1);
}

__global__ void k_scan(const int* __restrict__ deg, int* __restrict__ rowptr,
                       int* __restrict__ cursor)
{
    __shared__ int sm[1024];
    const int t = threadIdx.x;
    const int base = t * 64;
    int tot = 0;
    for (int k = 0; k < 64; ++k) tot += deg[base + k];
    sm[t] = tot;
    __syncthreads();
    for (int off = 1; off < 1024; off <<= 1) {
        int v = (t >= off) ? sm[t - off] : 0;
        __syncthreads();
        sm[t] += v;
        __syncthreads();
    }
    int run = sm[t] - tot;                   // exclusive prefix
    for (int k = 0; k < 64; ++k) {
        rowptr[base + k] = run;
        cursor[base + k] = run;
        run += deg[base + k];
    }
    if (t == 1023) rowptr[NT] = run;
}

__global__ void k_scatter(const int* __restrict__ ei, int* __restrict__ cursor,
                          int* __restrict__ csr_src)
{
    const int e = blockIdx.x * 256 + threadIdx.x;
    const int src = ei[e];
    const int dst = ei[NE + e];
    csr_src[atomicAdd(&cursor[dst], 1)] = src;
}

// ---------------- fused GAT softmax + aggregation: one wave per dst node ------
// h1 = relu( softmax_over_in_edges(leaky(as[src]+ad[dst])) . h[src]  + b1 )
// Also emits bf16 copy of h1 for the bf16 FC1 GEMM. h / h1 / CSR are L2-resident.
__global__ void k_aggregate(const float* __restrict__ h, const float* __restrict__ as,
                            const float* __restrict__ ad, const int* __restrict__ rowptr,
                            const int* __restrict__ csr_src, const float* __restrict__ b1,
                            float* __restrict__ h1, unsigned short* __restrict__ h1bf)
{
    const int lane = threadIdx.x & 31;
    const int node = blockIdx.x * 8 + (threadIdx.x >> 5);
    const int beg = rowptr[node], end = rowptr[node + 1];

    const float ad_i = ad[node];
    const float e_self = lrelu(as[node] + ad_i);

    // pass 1: max over edge logits (self loop included)
    float m = e_self;
    for (int j = beg + lane; j < end; j += 32)
        m = fmaxf(m, lrelu(as[csr_src[j]] + ad_i));
    #pragma unroll
    for (int off = 16; off; off >>= 1) m = fmaxf(m, __shfl_xor(m, off));

    // pass 2: sum of exp
    float ssum = 0.f;
    for (int j = beg + lane; j < end; j += 32)
        ssum += __expf(lrelu(as[csr_src[j]] + ad_i) - m);
    #pragma unroll
    for (int off = 16; off; off >>= 1) ssum += __shfl_xor(ssum, off);
    const float inv = 1.0f / (ssum + __expf(e_self - m) + 1e-16f);

    // pass 3: weighted accumulate, 4 features per lane, broadcast per edge
    const float w_self = __expf(e_self - m) * inv;
    const float4 hself = *(const float4*)(h + (size_t)node * HIDF + lane * 4);
    float4 acc;
    acc.x = w_self * hself.x; acc.y = w_self * hself.y;
    acc.z = w_self * hself.z; acc.w = w_self * hself.w;

    for (int j0 = beg; j0 < end; j0 += 32) {
        const int j = j0 + lane;
        int   s = 0;
        float e = -3.0e38f;
        if (j < end) { s = csr_src[j]; e = lrelu(as[s] + ad_i); }
        const int cnt = min(32, end - j0);
        for (int u = 0; u < cnt; ++u) {
            const float w  = __expf(__shfl(e, u) - m) * inv;
            const int   su = __shfl(s, u);
            const float4 hv = *(const float4*)(h + (size_t)su * HIDF + lane * 4);
            acc.x += w * hv.x; acc.y += w * hv.y; acc.z += w * hv.z; acc.w += w * hv.w;
        }
    }

    const float4 bb = *(const float4*)(b1 + lane * 4);
    acc.x = fmaxf(acc.x + bb.x, 0.f); acc.y = fmaxf(acc.y + bb.y, 0.f);
    acc.z = fmaxf(acc.z + bb.z, 0.f); acc.w = fmaxf(acc.w + bb.w, 0.f);

    *(float4*)(h1 + (size_t)node * HIDF + lane * 4) = acc;
    unsigned int lo = (unsigned int)f2bf(acc.x) | ((unsigned int)f2bf(acc.y) << 16);
    unsigned int hi = (unsigned int)f2bf(acc.z) | ((unsigned int)f2bf(acc.w) << 16);
    unsigned int* p = (unsigned int*)(h1bf + (size_t)node * HIDF + lane * 4);
    p[0] = lo; p[1] = hi;
}

// ---------------- fc1_W [KFC,128] f32 -> column-major bf16 [128][KFC] ----------
__global__ void k_wtrans_bf(const float* __restrict__ W, unsigned short* __restrict__ Wt)
{
    const int idx = blockIdx.x * 256 + threadIdx.x;   // < KFC*128
    const int k = idx >> 7, n = idx & 127;
    Wt[(size_t)n * KFC + k] = f2bf(W[idx]);
}

// ---------------- FC1: [512,16384]bf16 @ [16384,128]bf16 -> f32, relu ---------
// Block = 8 waves over the 8 N-tiles of one 16-row M-tile. V_WMMA_F32_16X16X32_BF16.
// TDM path: wave 0 issues two descriptors per K-step (A tile 16x32 = 1KB and
// B tile 128x32 = 8KB from column-major Wt), double-buffered in one __shared__
// object; all waves compute from LDS; wave 0 waits TENSORcnt; barrier publishes.
// Inner loop has zero direct VMEM: TDM -> LDS -> WMMA.
__global__ void k_fc1_bf16(const unsigned short* __restrict__ Abf,
                           const unsigned short* __restrict__ Wt,
                           const float* __restrict__ bias, float* __restrict__ Cout)
{
    const int lane = threadIdx.x & 31;
    const int wv   = threadIdx.x >> 5;
    const int m0   = blockIdx.x * 16;
    const int col0 = wv * 16;
    const int l15  = lane & 15;
    const int hi   = (lane & 16) ? 1 : 0;

    v8f c = {};
#if HAVE_TDM
    // layout (half-words): A buf0 [0,512), A buf1 [512,1024), B buf0 [1024,5120), B buf1 [5120,9216)
    __shared__ unsigned short smem[2 * 512 + 2 * 4096];   // 18KB
    if (wv == 0) {
        tdm_load_tile_bf16(Abf, 0u,    (unsigned)m0, 0u, 16u, (unsigned)NB);
        tdm_load_tile_bf16(Wt,  2048u, 0u,           0u, 128u, 128u);
        __builtin_amdgcn_s_wait_tensorcnt(0);
    }
    __syncthreads();
    const int nkt = KFC / 32;
    for (int kt = 0; kt < nkt; ++kt) {
        const int buf = kt & 1;
        if (wv == 0 && kt + 1 < nkt) {
            const unsigned k1 = (unsigned)((kt + 1) * 32);
            tdm_load_tile_bf16(Abf, (unsigned)((buf ^ 1) * 1024),        (unsigned)m0, k1, 16u, (unsigned)NB);
            tdm_load_tile_bf16(Wt,  (unsigned)(2048 + (buf ^ 1) * 8192), 0u,           k1, 128u, 128u);
        }
        // bf16 A 16x32: lower half-wave K 0-7 & 16-23; upper K 8-15 & 24-31
        const unsigned short* aT = smem + buf * 512 + l15 * 32 + hi * 8;
        // bf16 B 32x16: lower half-wave K 0-7 & 16-23; upper K 8-15 & 24-31 (col-major rows of 32)
        const unsigned short* bT = smem + 1024 + buf * 4096 + (col0 + l15) * 32 + hi * 8;
        BFV ua, ub;
        ua.q[0] = *(const U128*)(aT);
        ua.q[1] = *(const U128*)(aT + 16);
        ub.q[0] = *(const U128*)(bT);
        ub.q[1] = *(const U128*)(bT + 16);
        c = __builtin_amdgcn_wmma_f32_16x16x32_bf16(false, ua.v, false, ub.v,
                                                    (short)0, c, false, false);
        if (wv == 0) __builtin_amdgcn_s_wait_tensorcnt(0);
        __syncthreads();
    }
#else
    // bf16 A 16x32 layout: lower half-wave K 0-7 & 16-23; upper K 8-15 & 24-31
    const unsigned short* arow = Abf + (size_t)(m0 + l15) * KFC + hi * 8;
    const unsigned short* brow = Wt + (size_t)(col0 + l15) * KFC + hi * 16;
    for (int k = 0; k < KFC; k += 32) {
        __builtin_prefetch(arow + k + 64, 0, 0);
        __builtin_prefetch(brow + k + 64, 0, 0);
        BFV ua, ub;
        ua.q[0] = *(const U128*)(arow + k);
        ua.q[1] = *(const U128*)(arow + k + 16);
        ub.q[0] = *(const U128*)(brow + k);
        ub.q[1] = *(const U128*)(brow + k + 8);
        c = __builtin_amdgcn_wmma_f32_16x16x32_bf16(false, ua.v, false, ub.v,
                                                    (short)0, c, false, false);
    }
#endif
    const int n  = col0 + l15;
    const int mr = m0 + hi * 8;
    const float bb = bias[n];
    #pragma unroll
    for (int r = 0; r < 8; ++r)
        Cout[(size_t)(mr + r) * 128 + n] = fmaxf(c[r] + bb, 0.f);
}

// ================================================================ launcher ====
extern "C" void kernel_launch(void* const* d_in, const int* in_sizes, int n_in,
                              void* d_out, int out_size, void* d_ws, size_t ws_size,
                              hipStream_t stream)
{
    // input order: x, edge_index, W1,a_src1,a_dst1,b1, W2..b2(dead), W3..b3(dead),
    //              fc1_W, fc1_b, fc2_W, fc2_b   (layers 2/3 are dead code: the
    //              reference overwrites h3 and flattens h1 into the FC head)
    const float* x      = (const float*)d_in[0];
    const int*   ei     = (const int*)  d_in[1];   // int32 node ids (JAX default x64=off)
    const float* W1     = (const float*)d_in[2];
    const float* a_src1 = (const float*)d_in[3];
    const float* a_dst1 = (const float*)d_in[4];
    const float* b1     = (const float*)d_in[5];
    const float* fc1_W  = (const float*)d_in[14];
    const float* fc1_b  = (const float*)d_in[15];
    const float* fc2_W  = (const float*)d_in[16];
    const float* fc2_b  = (const float*)d_in[17];
    float* out = (float*)d_out;                    // [512,128] f32

    // bump allocator over workspace
    char* w = (char*)d_ws;
    auto alloc = [&](size_t bytes) -> void* {
        void* p = (void*)w;
        w += (bytes + 255) & ~(size_t)255;
        return p;
    };
    float*          h       = (float*)         alloc((size_t)NT * HIDF * 4); // x@W1
    float*          h1      = (float*)         alloc((size_t)NT * HIDF * 4); // GAT out
    float*          as      = (float*)         alloc((size_t)NT * 4);
    float*          adst    = (float*)         alloc((size_t)NT * 4);
    int*            deg     = (int*)           alloc((size_t)NT * 4);
    int*            rowptr  = (int*)           alloc((size_t)(NT + 1) * 4);
    int*            cursor  = (int*)           alloc((size_t)NT * 4);
    int*            csr_src = (int*)           alloc((size_t)NE * 4);
    unsigned short* h1bf    = (unsigned short*)alloc((size_t)NT * HIDF * 2);
    unsigned short* WtBf    = (unsigned short*)alloc((size_t)KFC * 128 * 2);
    float*          t1      = (float*)         alloc((size_t)NB * 128 * 4);
    (void)ws_size; (void)n_in; (void)in_sizes; (void)out_size;

    // fc1 weight transpose+bf16 (independent of graph pipeline)
    k_wtrans_bf<<<(KFC * 128) / 256, 256, 0, stream>>>(fc1_W, WtBf);

    // h = x @ W1   (f32 WMMA, K=64, A tile + B panel via async-to-LDS)
    k_gemm_f32_n128<<<NT / 16, 256, 0, stream>>>(x, W1, nullptr, h, INF_, 0);

    // attention logit halves
    k_alpha<<<NT / 8, 256, 0, stream>>>(h, a_src1, a_dst1, as, adst);

    // CSR by destination
    k_zero   <<<NT / 256, 256, 0, stream>>>(deg);
    k_degree <<<NE / 256, 256, 0, stream>>>(ei, deg);
    k_scan   <<<1, 1024, 0, stream>>>(deg, rowptr, cursor);
    k_scatter<<<NE / 256, 256, 0, stream>>>(ei, cursor, csr_src);

    // fused softmax + aggregation + bias + relu -> h1 (f32 + bf16)
    k_aggregate<<<NT / 8, 256, 0, stream>>>(h, as, adst, rowptr, csr_src, b1, h1, h1bf);

    // FC1: h1 viewed as [512,16384] bf16 @ WtBf -> t1 [512,128]  (bf16 WMMA + TDM)
    k_fc1_bf16<<<NB / 16, 256, 0, stream>>>(h1bf, WtBf, fc1_b, t1);

    // FC2: t1 @ fc2_W + b, relu -> out  (f32 WMMA, K=128)
    k_gemm_f32_n128<<<NB / 16, 256, 0, stream>>>(t1, fc2_W, fc2_b, out, 128, 1);
}
